// FlashGTA_87668872446002
// MI455X (gfx1250) — matrix operand order repaired and checked by reading, not compiled, not run on hardware
//
#include <hip/hip_runtime.h>
#include <stdint.h>

#define B_   2
#define L_   2048
#define D_   2048
#define H_   16
#define KV_  4
#define HD_  128
#define RD_  64
#define G_   4
#define MTOT (B_*L_)     // 4096

typedef __attribute__((ext_vector_type(16))) __bf16          v16bf;
typedef __attribute__((ext_vector_type(8)))  float           v8f;
typedef __attribute__((ext_vector_type(8)))  unsigned short  us8;

union FragU { struct { us8 lo; us8 hi; } p; v16bf v; };

static __device__ __forceinline__ v16bf mkfrag(us8 lo, us8 hi) {
    FragU u; u.p.lo = lo; u.p.hi = hi; return u.v;
}
static __device__ __forceinline__ unsigned short f2bf(float f) {
    unsigned u = __builtin_bit_cast(unsigned, f);
    u += 0x7fffu + ((u >> 16) & 1u);          // round to nearest even
    return (unsigned short)(u >> 16);
}
static __device__ __forceinline__ float bf2f(unsigned short h) {
    unsigned u = ((unsigned)h) << 16;
    return __builtin_bit_cast(float, u);
}

// Async global -> LDS copy, 16 bytes per lane. LDS byte offset = low 32 bits
// of the flat pointer (ISA 10.2: LDS aperture truncates to addr[31:0]).
static __device__ __forceinline__ void async_load_b128(void* lds_ptr, const void* gptr) {
    unsigned loff = (unsigned)(uintptr_t)lds_ptr;
    unsigned long long ga = (unsigned long long)(uintptr_t)gptr;
    asm volatile("global_load_async_to_lds_b128 %0, %1, off"
                 :: "v"(loff), "v"(ga) : "memory");
}
static __device__ __forceinline__ void wait_async0() {
    asm volatile("s_wait_asynccnt 0x0" ::: "memory");
}

// ---------------------------------------------------------------- convert
__global__ void f32_to_bf16_k(const float* __restrict__ in,
                              unsigned short* __restrict__ out, int n) {
    int i = blockIdx.x * blockDim.x + threadIdx.x;
    int stride = gridDim.x * blockDim.x;
    for (; i < n; i += stride) out[i] = f2bf(in[i]);
}

// ---------------------------------------------------------------- GEMM
// C[M,N] = A[M,K](bf16,row-major) * W[N,K](bf16,row-major)^T + bias[N]
// 256 threads / 8 waves; block tile 128(M) x 64(N); each wave owns a 16x64
// strip (1 A-frag reused over 4 B-frags = 4 WMMA per barrier). K chunks of 32,
// double-buffered async global->LDS staging.
template <bool OUT_F32>
__global__ __launch_bounds__(256)
void gemm_bf16_wmma(const unsigned short* __restrict__ A,
                    const unsigned short* __restrict__ W,
                    const float* __restrict__ bias,
                    void* __restrict__ Cout,
                    int M, int N, int K) {
    __shared__ __align__(16) unsigned short As[2][128][40];
    __shared__ __align__(16) unsigned short Ws[2][64][40];

    const int tid  = threadIdx.x;
    const int lane = tid & 31, ln = lane & 15, hi = lane >> 4;
    const int w    = tid >> 5;
    const int m_sub = w << 4;                  // 0..112
    const int mtile = blockIdx.y << 7;
    const int ntile = blockIdx.x << 6;

    const int arow = tid >> 1, acol = (tid & 1) << 4;   // 128x32 A stage
    const int wrow = tid >> 2, wcol = (tid & 3) << 3;   // 64x32  W stage

    auto stage = [&](int buf, int k0) {
        const unsigned short* ga = &A[(size_t)(mtile + arow) * K + k0 + acol];
        async_load_b128(&As[buf][arow][acol],     ga);
        async_load_b128(&As[buf][arow][acol + 8], ga + 8);
        const unsigned short* gw = &W[(size_t)(ntile + wrow) * K + k0 + wcol];
        async_load_b128(&Ws[buf][wrow][wcol], gw);
    };

    v8f acc[4];
#pragma unroll
    for (int nt = 0; nt < 4; ++nt) acc[nt] = 0.f;

    const int nCh = K >> 5;
    stage(0, 0);
    int cur = 0;
    for (int ch = 0; ch < nCh; ++ch) {
        wait_async0();
        __syncthreads();
        if (ch + 1 < nCh) stage(cur ^ 1, (ch + 1) << 5);

        us8 alo = *(const us8*)&As[cur][m_sub + ln][hi * 8];
        us8 ahi = *(const us8*)&As[cur][m_sub + ln][hi * 8 + 16];
        v16bf a = mkfrag(alo, ahi);
#pragma unroll
        for (int nt = 0; nt < 4; ++nt) {
            us8 blo = *(const us8*)&Ws[cur][nt * 16 + ln][hi * 8];
            us8 bhi = *(const us8*)&Ws[cur][nt * 16 + ln][hi * 8 + 16];
            v16bf b = mkfrag(blo, bhi);
            acc[nt] = __builtin_amdgcn_wmma_f32_16x16x32_bf16(
                          false, a, false, b, (short)0, acc[nt], false, false);
        }
        cur ^= 1;
    }

#pragma unroll
    for (int nt = 0; nt < 4; ++nt) {
        const int ng = ntile + nt * 16 + ln;
        const float bv = bias[ng];
#pragma unroll
        for (int r = 0; r < 8; ++r) {
            int mg = mtile + m_sub + r + 8 * hi;
            float v = acc[nt][r] + bv;
            if (OUT_F32) ((float*)Cout)[(size_t)mg * N + ng] = v;
            else ((unsigned short*)Cout)[(size_t)mg * N + ng] = f2bf(v);
        }
    }
}

// ---------------------------------------------------------------- RoPE + k_states
__global__ __launch_bounds__(256)
void rope_build_k(unsigned short* __restrict__ q,
                  const unsigned short* __restrict__ kv,
                  const unsigned short* __restrict__ kr,
                  unsigned short* __restrict__ kst,
                  const float* __restrict__ cosp,
                  const float* __restrict__ sinp) {
    const int t = blockIdx.x;
    const int tid = threadIdx.x;
    const float qscale = 0.08838834764831845f;   // 1/sqrt(128)
    const size_t qbase = (size_t)t * (H_ * HD_);
    const size_t cbase = (size_t)t * RD_;

    float nv[8];
#pragma unroll
    for (int j = 0; j < 8; ++j) {
        int e = tid + 256 * j;
        int d = e & (HD_ - 1);
        float val = bf2f(q[qbase + e]);
        float outv = val;
        if (d >= HD_ - RD_) {
            int r = d - (HD_ - RD_);
            int pe = (r < RD_ / 2) ? (e + RD_ / 2) : (e - RD_ / 2);
            float pv = bf2f(q[qbase + pe]);
            outv = val * cosp[cbase + r] +
                   ((r < RD_ / 2) ? -pv : pv) * sinp[cbase + r];
        }
        nv[j] = outv * qscale;
    }
    __syncthreads();
#pragma unroll
    for (int j = 0; j < 8; ++j) q[qbase + tid + 256 * j] = f2bf(nv[j]);

    const size_t kvb = (size_t)t * (KV_ * HD_);
    const size_t krb = (size_t)t * RD_;
#pragma unroll
    for (int j = 0; j < 2; ++j) {
        int e = tid + 256 * j;
        int d = e & (HD_ - 1);
        unsigned short o;
        if (d < HD_ - RD_) {
            o = kv[kvb + e];
        } else {
            int r = d - (HD_ - RD_);
            float val = bf2f(kr[krb + r]);
            int pr = (r < RD_ / 2) ? r + RD_ / 2 : r - RD_ / 2;
            float pv = bf2f(kr[krb + pr]);
            o = f2bf(val * cosp[cbase + r] +
                     ((r < RD_ / 2) ? -pv : pv) * sinp[cbase + r]);
        }
        kst[kvb + e] = o;
    }
}

// ---------------------------------------------------------------- flash attention
// grid: (L/128, H, B); block 256 = 8 waves; wave owns 16 query rows.
__global__ __launch_bounds__(256)
void flash_attn_wmma(const unsigned short* __restrict__ q,    // (MTOT,H,HD) pre-scaled
                     const unsigned short* __restrict__ kst,  // (MTOT,KV,HD)
                     const unsigned short* __restrict__ vst,  // (MTOT,KV,HD)
                     unsigned short* __restrict__ att) {      // (MTOT,H*HD)
    __shared__ __align__(16) unsigned short Ks[32][136];   // keys x HD (padded)
    __shared__ __align__(16) unsigned short Vt[128][40];   // HD x keys (transposed)
    __shared__ __align__(16) unsigned short Pb[8][16][32]; // per-wave P tile

    const int tid  = threadIdx.x;
    const int lane = tid & 31, ln = lane & 15, hi = lane >> 4;
    const int w    = tid >> 5;
    const int h    = blockIdx.y;
    const int kvh  = h / G_;
    const int b    = blockIdx.z;
    const int qblock0 = blockIdx.x * 128;
    const int qw0     = qblock0 + w * 16;

    v16bf qf[4];
    {
        const size_t qrow = ((size_t)(b * L_ + qw0 + ln) * H_ + h) * HD_;
#pragma unroll
        for (int kc = 0; kc < 4; ++kc) {
            us8 lo  = *(const us8*)&q[qrow + kc * 32 + hi * 8];
            us8 hi8 = *(const us8*)&q[qrow + kc * 32 + hi * 8 + 16];
            qf[kc] = mkfrag(lo, hi8);
        }
    }

    v8f acc[8];
#pragma unroll
    for (int i = 0; i < 8; ++i) acc[i] = 0.f;
    float m_i[8], l_i[8];
#pragma unroll
    for (int r = 0; r < 8; ++r) { m_i[r] = -1e30f; l_i[r] = 0.f; }

    const int key = tid >> 3, d0 = (tid & 7) << 4;
    const int nCh = blockIdx.x * 4 + 4;   // causal: keys up to qblock0+127

    for (int ch = 0; ch < nCh; ++ch) {
        const int s0 = ch * 32;
        {   // stage K via async-to-LDS; V via VGPR transpose
            size_t g = ((size_t)(b * L_ + s0 + key) * KV_ + kvh) * HD_ + d0;
            async_load_b128(&Ks[key][d0],     &kst[g]);
            async_load_b128(&Ks[key][d0 + 8], &kst[g + 8]);
            us8 v0v = *(const us8*)&vst[g];
            us8 v1v = *(const us8*)&vst[g + 8];
#pragma unroll
            for (int j = 0; j < 8; ++j) Vt[d0 + j][key]     = v0v[j];
#pragma unroll
            for (int j = 0; j < 8; ++j) Vt[d0 + 8 + j][key] = v1v[j];
        }
        wait_async0();
        __syncthreads();

        if (s0 <= qw0 + 15) {
            v8f st[2];
#pragma unroll
            for (int nt = 0; nt < 2; ++nt) {
                v8f s = 0.f;
#pragma unroll
                for (int kc = 0; kc < 4; ++kc) {
                    us8 blo = *(const us8*)&Ks[nt * 16 + ln][kc * 32 + hi * 8];
                    us8 bhi = *(const us8*)&Ks[nt * 16 + ln][kc * 32 + hi * 8 + 16];
                    v16bf bf = mkfrag(blo, bhi);
                    s = __builtin_amdgcn_wmma_f32_16x16x32_bf16(
                            false, qf[kc], false, bf, (short)0, s, false, false);
                }
                int colg = s0 + nt * 16 + ln;
#pragma unroll
                for (int r = 0; r < 8; ++r)
                    if (colg > qw0 + r + 8 * hi) s[r] = -1e30f;
                st[nt] = s;
            }

            float mloc[8], sc[8], su[8];
#pragma unroll
            for (int r = 0; r < 8; ++r) mloc[r] = fmaxf(st[0][r], st[1][r]);
#pragma unroll
            for (int msk = 1; msk <= 8; msk <<= 1)
#pragma unroll
                for (int r = 0; r < 8; ++r)
                    mloc[r] = fmaxf(mloc[r], __shfl_xor(mloc[r], msk, 32));
#pragma unroll
            for (int r = 0; r < 8; ++r) {
                float nm = fmaxf(m_i[r], mloc[r]);
                sc[r] = __expf(m_i[r] - nm);
                m_i[r] = nm;
            }
#pragma unroll
            for (int r = 0; r < 8; ++r) {
                st[0][r] = __expf(st[0][r] - m_i[r]);
                st[1][r] = __expf(st[1][r] - m_i[r]);
                su[r] = st[0][r] + st[1][r];
            }
#pragma unroll
            for (int msk = 1; msk <= 8; msk <<= 1)
#pragma unroll
                for (int r = 0; r < 8; ++r)
                    su[r] += __shfl_xor(su[r], msk, 32);
#pragma unroll
            for (int r = 0; r < 8; ++r) l_i[r] = l_i[r] * sc[r] + su[r];
#pragma unroll
            for (int dt = 0; dt < 8; ++dt)
#pragma unroll
                for (int r = 0; r < 8; ++r) acc[dt][r] *= sc[r];

            // C-layout P -> per-wave LDS -> A-layout fragment
#pragma unroll
            for (int nt = 0; nt < 2; ++nt)
#pragma unroll
                for (int r = 0; r < 8; ++r)
                    Pb[w][r + 8 * hi][nt * 16 + ln] = f2bf(st[nt][r]);
            asm volatile("s_wait_dscnt 0" ::: "memory");   // cross-lane LDS dep

            us8 plo = *(const us8*)&Pb[w][ln][hi * 8];
            us8 phi = *(const us8*)&Pb[w][ln][hi * 8 + 16];
            v16bf pf = mkfrag(plo, phi);
#pragma unroll
            for (int dt = 0; dt < 8; ++dt) {
                us8 vlo = *(const us8*)&Vt[dt * 16 + ln][hi * 8];
                us8 vhi = *(const us8*)&Vt[dt * 16 + ln][hi * 8 + 16];
                v16bf vf = mkfrag(vlo, vhi);
                acc[dt] = __builtin_amdgcn_wmma_f32_16x16x32_bf16(
                              false, pf, false, vf, (short)0, acc[dt], false, false);
            }
        }
        __syncthreads();
    }

#pragma unroll
    for (int dt = 0; dt < 8; ++dt)
#pragma unroll
        for (int r = 0; r < 8; ++r) {
            int mg = qw0 + r + 8 * hi;
            float o = acc[dt][r] / l_i[r];
            att[(size_t)(b * L_ + mg) * (H_ * HD_) + h * HD_ + dt * 16 + ln] =
                f2bf(o);
        }
}

// ---------------------------------------------------------------- launcher
extern "C" void kernel_launch(void* const* d_in, const int* in_sizes, int n_in,
                              void* d_out, int out_size, void* d_ws, size_t ws_size,
                              hipStream_t stream) {
    (void)in_sizes; (void)n_in; (void)out_size; (void)ws_size;
    const float* x    = (const float*)d_in[0];
    const float* cosp = (const float*)d_in[1];
    const float* sinp = (const float*)d_in[2];
    const float* Wq   = (const float*)d_in[3];
    const float* bq   = (const float*)d_in[4];
    const float* Wkv  = (const float*)d_in[5];
    const float* bkv  = (const float*)d_in[6];
    const float* Wrk  = (const float*)d_in[7];
    const float* brk  = (const float*)d_in[8];
    const float* Wo   = (const float*)d_in[9];
    const float* bo   = (const float*)d_in[10];

    unsigned short* ws = (unsigned short*)d_ws;
    size_t off = 0;
    auto alloc = [&](size_t n) { unsigned short* p = ws + off; off += n; return p; };
    unsigned short* xb   = alloc((size_t)MTOT * D_);
    unsigned short* wqb  = alloc((size_t)(H_ * HD_) * D_);
    unsigned short* wkvb = alloc((size_t)(KV_ * HD_) * D_);
    unsigned short* wrkb = alloc((size_t)RD_ * D_);
    unsigned short* wob  = alloc((size_t)D_ * (H_ * HD_));
    unsigned short* qp   = alloc((size_t)MTOT * (H_ * HD_));
    unsigned short* kvp  = alloc((size_t)MTOT * (KV_ * HD_));
    unsigned short* krp  = alloc((size_t)MTOT * RD_);
    unsigned short* kstp = alloc((size_t)MTOT * (KV_ * HD_));
    unsigned short* attp = alloc((size_t)MTOT * (H_ * HD_));

    auto cvt = [&](const float* src, unsigned short* dst, size_t n) {
        int blocks = (int)((n + 1023) / 1024);
        if (blocks > 4096) blocks = 4096;
        f32_to_bf16_k<<<blocks, 256, 0, stream>>>(src, dst, (int)n);
    };
    cvt(x,   xb,   (size_t)MTOT * D_);
    cvt(Wq,  wqb,  (size_t)(H_ * HD_) * D_);
    cvt(Wkv, wkvb, (size_t)(KV_ * HD_) * D_);
    cvt(Wrk, wrkb, (size_t)RD_ * D_);
    cvt(Wo,  wob,  (size_t)D_ * (H_ * HD_));

    // projections
    gemm_bf16_wmma<false><<<dim3((H_ * HD_) / 64, MTOT / 128), 256, 0, stream>>>(
        xb, wqb, bq, qp, MTOT, H_ * HD_, D_);
    gemm_bf16_wmma<false><<<dim3((KV_ * HD_) / 64, MTOT / 128), 256, 0, stream>>>(
        xb, wkvb, bkv, kvp, MTOT, KV_ * HD_, D_);
    gemm_bf16_wmma<false><<<dim3(RD_ / 64, MTOT / 128), 256, 0, stream>>>(
        xb, wrkb, brk, krp, MTOT, RD_, D_);

    // rope + k_states
    rope_build_k<<<MTOT, 256, 0, stream>>>(qp, kvp, krp, kstp, cosp, sinp);

    // grouped causal flash attention
    flash_attn_wmma<<<dim3(L_ / 128, H_, B_), 256, 0, stream>>>(qp, kstp, kvp, attp);

    // output projection (f32 out)
    gemm_bf16_wmma<true><<<dim3(D_ / 64, MTOT / 128), 256, 0, stream>>>(
        attp, wob, bo, d_out, MTOT, D_, H_ * HD_);
}
